// Transformer_30571577212950
// MI455X (gfx1250) — matrix operand order repaired and checked by reading, not compiled, not run on hardware
//
#include <hip/hip_runtime.h>
#include <hip/hip_bf16.h>
#include <math.h>

#define D_ 1024
#define H_ 8
#define DH_ 128
#define L_ 6
#define V_ 32000
#define B_ 4
#define N_ 512
#define M_ 512
#define EPS_ 1e-5f
#define SLOPE_ 0.01f

typedef __attribute__((ext_vector_type(16))) _Float16 v16h;
typedef __attribute__((ext_vector_type(8)))  _Float16 v8h;
typedef __attribute__((ext_vector_type(8)))  float    v8f;

#define LDSS 48  // LDS row stride in halves (96B rows; 16B/32B aligned chunks)

// ---------------------------------------------------------------------------
// WMMA GEMM, fp16 operands, B always in "transposed" layout (cols x K, K-contig).
//   C(f32) = act(alpha * A @ B^T + bias) + residual      (flags 2 bias, 4 leaky, 8 res)
//   or f16 output:    flags 16 -> Ch[row*ldch+col]       (plain f16)
//                     flags 32 -> Ch[col*ldch+row]       (transposed f16)
// Block 128 thr = 4 waves; block tile 128x64; wave tile 32x64 (8 WMMA / K-step).
// Double-buffered LDS staging via GLOBAL_LOAD_ASYNC_TO_LDS_B128 (ASYNCcnt).
// Batched over blockIdx.z with (outer, inner) stride decomposition.
// ---------------------------------------------------------------------------
__global__ void __launch_bounds__(128)
gemm_wmma_k(const _Float16* __restrict__ A, const _Float16* __restrict__ Bm,
            const float* __restrict__ bias, const float* __restrict__ res,
            float* __restrict__ C, _Float16* __restrict__ Ch,
            int K, int lda, int ldb, int ldc, int ldch,
            long long sAo, long long sAi, long long sBo, long long sBi,
            long long sCo, long long sCi, int innerN,
            float alpha, int flags)
{
  __shared__ _Float16 As[2][128 * LDSS];
  __shared__ _Float16 Bs[2][64 * LDSS];

  const int z  = blockIdx.z;
  const int zo = z / innerN;
  const int zi = z - zo * innerN;
  A  += zo * sAo + zi * sAi;
  Bm += zo * sBo + zi * sBi;
  const long long coff = zo * sCo + zi * sCi;

  const int tile_m = blockIdx.x * 128;
  const int tile_n = blockIdx.y * 64;
  const int t    = threadIdx.x;
  const int wave = t >> 5;
  const int lane = t & 31;
  const int l16  = lane & 15;
  const int hi   = lane >> 4;     // half-wave select
  const int kbA  = hi * 16;       // A fragment K sub-base (halves)
  const int kbB  = hi * 16;       // B fragment K sub-base (halves)

  // issue async global->LDS copies for one K-step (A: 128x32, B: 64x32 halves)
  auto stage = [&](int buf, int k0) {
#pragma unroll
    for (int i = 0; i < 4; i++) {               // A: 512 x 16B chunks / 128 thr
      const int chunk = t + i * 128;
      const int r  = chunk >> 2;
      const int c8 = (chunk & 3) * 8;
      const void* g = (const void*)(A + (long long)(tile_m + r) * lda + k0 + c8);
      const unsigned l = (unsigned)(size_t)&As[buf][r * LDSS + c8];
      asm volatile("global_load_async_to_lds_b128 %0, %1, off" :: "v"(l), "v"(g) : "memory");
    }
#pragma unroll
    for (int i = 0; i < 2; i++) {               // B: 256 x 16B chunks / 128 thr
      const int chunk = t + i * 128;
      const int n  = chunk >> 2;
      const int c8 = (chunk & 3) * 8;
      const void* g = (const void*)(Bm + (long long)(tile_n + n) * ldb + k0 + c8);
      const unsigned l = (unsigned)(size_t)&Bs[buf][n * LDSS + c8];
      asm volatile("global_load_async_to_lds_b128 %0, %1, off" :: "v"(l), "v"(g) : "memory");
    }
  };

  v8f acc[2][4] = {};

  stage(0, 0);
  asm volatile("s_wait_asynccnt 0x0" ::: "memory");
  __syncthreads();

  int buf = 0;
  for (int k0 = 0; k0 < K; k0 += 32) {
    if (k0 + 32 < K) stage(buf ^ 1, k0 + 32);   // prefetch overlaps with WMMAs

    // A fragments (contiguous K layout: lane row, K {kb..kb+15}, matches B)
    v16h a0, a1;
    {
      const int r0 = (wave * 32 + l16) * LDSS;
      const v8h lo0 = *(const v8h*)&As[buf][r0 + kbA];
      const v8h hi0 = *(const v8h*)&As[buf][r0 + kbA + 8];
      const int r1 = r0 + 16 * LDSS;
      const v8h lo1 = *(const v8h*)&As[buf][r1 + kbA];
      const v8h hi1 = *(const v8h*)&As[buf][r1 + kbA + 8];
#pragma unroll
      for (int i = 0; i < 8; i++) {
        a0[i] = lo0[i]; a0[i + 8] = hi0[i];
        a1[i] = lo1[i]; a1[i + 8] = hi1[i];
      }
    }
#pragma unroll
    for (int j = 0; j < 4; j++) {
      const v16h b = *(const v16h*)&Bs[buf][(j * 16 + l16) * LDSS + kbB];
      acc[0][j] = __builtin_amdgcn_wmma_f32_16x16x32_f16(false, a0, false, b, (short)0, acc[0][j], false, false);
      acc[1][j] = __builtin_amdgcn_wmma_f32_16x16x32_f16(false, a1, false, b, (short)0, acc[1][j], false, false);
    }

    asm volatile("s_wait_asynccnt 0x0" ::: "memory");
    __syncthreads();
    buf ^= 1;
  }

  // epilogue: C layout = VGPR e -> row e (lanes 0-15) / e+8 (lanes 16-31)
  float* Cp = C ? C + coff : nullptr;
  _Float16* Chp = Ch ? Ch + coff : nullptr;
  const float* resp = res ? res + coff : nullptr;
#pragma unroll
  for (int f = 0; f < 2; f++) {
    const int rbase = tile_m + wave * 32 + f * 16 + hi * 8;
#pragma unroll
    for (int j = 0; j < 4; j++) {
      const int col = tile_n + j * 16 + l16;
#pragma unroll
      for (int e = 0; e < 8; e++) {
        float vv = acc[f][j][e] * alpha;
        if (flags & 2) vv += bias[col];
        if (flags & 4) vv = vv > 0.0f ? vv : vv * SLOPE_;
        const int row = rbase + e;
        if (flags & 16) {
          Chp[(long long)row * ldch + col] = (_Float16)vv;
        } else if (flags & 32) {
          Chp[(long long)col * ldch + row] = (_Float16)vv;
        } else {
          const long long idx = (long long)row * ldc + col;
          if (flags & 8) vv += resp[idx];
          Cp[idx] = vv;
        }
      }
    }
  }
}

// ---------------------------------------------------------------------------
// Weight transpose + f32->f16 convert:  in (K x N) f32  ->  out (N x K) f16
// ---------------------------------------------------------------------------
__global__ void __launch_bounds__(256)
transpose_cvt_k(const float* __restrict__ in, _Float16* __restrict__ out, int K, int N)
{
  __shared__ float tile[32][33];
  const int k0 = blockIdx.x * 32;
  const int n0 = blockIdx.y * 32;
  const int tx = threadIdx.x & 31;
  const int ty = threadIdx.x >> 5;  // 0..7
  for (int j = ty; j < 32; j += 8)
    tile[j][tx] = in[(long long)(k0 + j) * N + n0 + tx];
  __syncthreads();
  for (int j = ty; j < 32; j += 8)
    out[(long long)(n0 + j) * K + k0 + tx] = (_Float16)tile[tx][j];
}

// ---------------------------------------------------------------------------
// LayerNorm: reads f32, writes f16 (GEMM operand) and optionally f32 (residual)
// ---------------------------------------------------------------------------
__global__ void __launch_bounds__(256)
layernorm_k(const float* __restrict__ x, const float* __restrict__ g,
            const float* __restrict__ b, _Float16* __restrict__ yh,
            float* __restrict__ yf, int Dn)
{
  __shared__ float s1[256];
  __shared__ float s2[256];
  const int row = blockIdx.x;
  const int t = threadIdx.x;
  const float* xr = x + (long long)row * Dn;
  float s = 0.f, ss = 0.f;
  for (int i = t; i < Dn; i += 256) { const float vv = xr[i]; s += vv; ss += vv * vv; }
  s1[t] = s; s2[t] = ss;
  __syncthreads();
  for (int st = 128; st > 0; st >>= 1) {
    if (t < st) { s1[t] += s1[t + st]; s2[t] += s2[t + st]; }
    __syncthreads();
  }
  const float mu  = s1[0] / (float)Dn;
  const float var = s2[0] / (float)Dn - mu * mu;
  const float r = rsqrtf(var + EPS_);
  _Float16* yhr = yh + (long long)row * Dn;
  float* yfr = yf ? yf + (long long)row * Dn : nullptr;
  for (int i = t; i < Dn; i += 256) {
    const float o = (xr[i] - mu) * r * g[i] + b[i];
    yhr[i] = (_Float16)o;
    if (yfr) yfr[i] = o;
  }
}

// ---------------------------------------------------------------------------
// Token embedding + sinusoidal positional encoding (f32 residual stream)
// ---------------------------------------------------------------------------
__global__ void __launch_bounds__(256)
embed_k(const int* __restrict__ tok, const float* __restrict__ emb,
        float* __restrict__ out, int Dn, int seqN)
{
  const int row = blockIdx.x;
  const int pos = row % seqN;
  const int tk  = tok[row];
  const int t = threadIdx.x;
  const float* er = emb + (long long)tk * Dn;
  float* orow = out + (long long)row * Dn;
  for (int d = t; d < Dn; d += 256) {
    const int i = d >> 1;
    const float ang = (float)pos * expf(-((float)i / (float)Dn) * 9.210340371976184f);
    const float pe = (d & 1) ? cosf(ang) : sinf(ang);
    orow[d] = er[d] + pe;
  }
}

// ---------------------------------------------------------------------------
// Masked softmax: reads f32 scores, writes f16 probs.
// mode 0: key-padding (n < xlen[b]); mode 1: causal (n <= m). Grid (M, B*H).
// ---------------------------------------------------------------------------
__global__ void __launch_bounds__(256)
attn_softmax_k(const float* __restrict__ s, _Float16* __restrict__ p,
               const int* __restrict__ xlen, int Ncols, int Hh, int mode)
{
  __shared__ float red[256];
  const int m = blockIdx.x;
  const int z = blockIdx.y;
  const int bidx = z / Hh;
  const long long roff = ((long long)z * gridDim.x + m) * Ncols;
  const float* row = s + roff;
  _Float16* prow = p + roff;
  const int t = threadIdx.x;
  const int limit = (mode == 0) ? xlen[bidx] : (m + 1);

  float mx = -INFINITY;
  for (int n = t; n < Ncols; n += 256)
    mx = fmaxf(mx, (n < limit) ? row[n] : -INFINITY);
  red[t] = mx; __syncthreads();
  for (int st = 128; st > 0; st >>= 1) { if (t < st) red[t] = fmaxf(red[t], red[t + st]); __syncthreads(); }
  mx = red[0];
  __syncthreads();

  float sum = 0.f;
  for (int n = t; n < Ncols; n += 256)
    sum += (n < limit) ? expf(row[n] - mx) : 0.f;
  red[t] = sum; __syncthreads();
  for (int st = 128; st > 0; st >>= 1) { if (t < st) red[t] += red[t + st]; __syncthreads(); }
  const float inv = 1.f / red[0];

  for (int n = t; n < Ncols; n += 256)
    prow[n] = (_Float16)((n < limit) ? expf(row[n] - mx) * inv : 0.f);
}

// ---------------------------------------------------------------------------
// In-place log_softmax over last dim
// ---------------------------------------------------------------------------
__global__ void __launch_bounds__(256)
log_softmax_k(float* __restrict__ x, int Ncols)
{
  __shared__ float red[256];
  const int row = blockIdx.x;
  float* xr = x + (long long)row * Ncols;
  const int t = threadIdx.x;
  float mx = -INFINITY;
  for (int i = t; i < Ncols; i += 256) mx = fmaxf(mx, xr[i]);
  red[t] = mx; __syncthreads();
  for (int st = 128; st > 0; st >>= 1) { if (t < st) red[t] = fmaxf(red[t], red[t + st]); __syncthreads(); }
  mx = red[0]; __syncthreads();
  float sum = 0.f;
  for (int i = t; i < Ncols; i += 256) sum += expf(xr[i] - mx);
  red[t] = sum; __syncthreads();
  for (int st = 128; st > 0; st >>= 1) { if (t < st) red[t] += red[t + st]; __syncthreads(); }
  const float lse = mx + logf(red[0]);
  for (int i = t; i < Ncols; i += 256) xr[i] = xr[i] - lse;
}

// ---------------------------------------------------------------------------
// Host-side helpers
// ---------------------------------------------------------------------------
static inline void gemmF(hipStream_t st,
    const _Float16* A, const _Float16* B, const float* bias, const float* res,
    float* C, _Float16* Ch,
    int rows, int cols, int K, int lda, int ldb, int ldc, int ldch,
    int batches, int innerN,
    long long sAo, long long sAi, long long sBo, long long sBi,
    long long sCo, long long sCi, float alpha, int flags)
{
  dim3 grid(rows / 128, cols / 64, batches);
  gemm_wmma_k<<<grid, 128, 0, st>>>(A, B, bias, res, C, Ch, K, lda, ldb, ldc, ldch,
                                    sAo, sAi, sBo, sBi, sCo, sCi, innerN, alpha, flags);
}

static inline void trcvt(hipStream_t st, const float* W, _Float16* WT, int K, int N)
{
  transpose_cvt_k<<<dim3(K / 32, N / 32), 256, 0, st>>>(W, WT, K, N);
}

// Full MHA: outF = (softmax(zq Wq (zkv Wk)^T / sqrt(dh) + mask) zkv Wv) Wo + residF
static void attention(hipStream_t st, const _Float16* zqh, const _Float16* zkvh,
                      const float* W, const int* xlen, int maskmode,
                      const float* residF, float* outF,
                      _Float16* qh, _Float16* kh, _Float16* vT,
                      float* sc, _Float16* probs, _Float16* oh, _Float16* wT)
{
  const long long DD = (long long)D_ * D_;
  const int R = B_ * N_;
  // projections (weights transposed+converted to f16 on the fly)
  trcvt(st, W + 0 * DD, wT, D_, D_);
  gemmF(st, zqh, wT, nullptr, nullptr, nullptr, qh, R, D_, D_, D_, D_, 0, D_,
        1, 1, 0, 0, 0, 0, 0, 0, 1.f, 16);
  trcvt(st, W + 1 * DD, wT, D_, D_);
  gemmF(st, zkvh, wT, nullptr, nullptr, nullptr, kh, R, D_, D_, D_, D_, 0, D_,
        1, 1, 0, 0, 0, 0, 0, 0, 1.f, 16);
  trcvt(st, W + 2 * DD, wT, D_, D_);
  gemmF(st, zkvh, wT, nullptr, nullptr, nullptr, vT, R, D_, D_, D_, D_, 0, R,
        1, 1, 0, 0, 0, 0, 0, 0, 1.f, 32);   // V written transposed (D x R)
  // scores[b,h] = q_bh @ k_bh^T / sqrt(dh), f32 out
  gemmF(st, qh, kh, nullptr, nullptr, sc, nullptr, M_, N_, DH_, D_, D_, N_, 0,
        B_ * H_, H_,
        (long long)M_ * D_, DH_,
        (long long)N_ * D_, DH_,
        (long long)H_ * M_ * N_, (long long)M_ * N_,
        0.08838834764831845f, 0);
  attn_softmax_k<<<dim3(M_, B_ * H_), 256, 0, st>>>(sc, probs, xlen, N_, H_, maskmode);
  // o[b,h] = probs @ v_bh  (uses transposed V as BT operand)
  gemmF(st, probs, vT, nullptr, nullptr, nullptr, oh, M_, DH_, N_, N_, R, 0, D_,
        B_ * H_, H_,
        (long long)H_ * M_ * N_, (long long)M_ * N_,
        (long long)N_, (long long)DH_ * R,
        (long long)M_ * D_, DH_,
        1.f, 16);
  // out = o @ Wo + residF (f32)
  trcvt(st, W + 3 * DD, wT, D_, D_);
  gemmF(st, oh, wT, nullptr, residF, outF, nullptr, R, D_, D_, D_, D_, D_, 0,
        1, 1, 0, 0, 0, 0, 0, 0, 1.f, 8);
}

extern "C" void kernel_launch(void* const* d_in, const int* in_sizes, int n_in,
                              void* d_out, int out_size, void* d_ws, size_t ws_size,
                              hipStream_t stream)
{
  (void)in_sizes; (void)n_in; (void)out_size; (void)ws_size;

  const int*   x_tokens = (const int*)d_in[0];
  const int*   x_len    = (const int*)d_in[1];
  const int*   y_tokens = (const int*)d_in[2];
  const float* emb_enc  = (const float*)d_in[3];
  const float* emb_dec  = (const float*)d_in[4];
  const float* enc_W    = (const float*)d_in[5];
  const float* enc_ln_g = (const float*)d_in[6];
  const float* enc_ln_b = (const float*)d_in[7];
  const float* enc_W1   = (const float*)d_in[8];
  const float* enc_b1   = (const float*)d_in[9];
  const float* enc_W2   = (const float*)d_in[10];
  const float* enc_b2   = (const float*)d_in[11];
  const float* dec_mW   = (const float*)d_in[12];
  const float* dec_cW   = (const float*)d_in[13];
  const float* dec_ln_g = (const float*)d_in[14];
  const float* dec_ln_b = (const float*)d_in[15];
  const float* dec_W1   = (const float*)d_in[16];
  const float* dec_b1   = (const float*)d_in[17];
  const float* dec_W2   = (const float*)d_in[18];
  const float* dec_b2   = (const float*)d_in[19];
  const float* gen_ln_g = (const float*)d_in[20];
  const float* gen_ln_b = (const float*)d_in[21];
  const float* gen_W    = (const float*)d_in[22];
  const float* gen_b    = (const float*)d_in[23];
  float* out = (float*)d_out;

  // ---- workspace carve-up ----
  const size_t TACT = (size_t)(B_ * N_) * D_;       // 2,097,152 elems
  const size_t TSC  = (size_t)B_ * H_ * M_ * N_;    // 8,388,608 elems
  const size_t TMID = (size_t)(B_ * N_) * 4 * D_;   // 8,388,608 elems
  const size_t TWT  = (size_t)V_ * D_;              // 32,768,000 elems
  char* p = (char*)d_ws;
  auto alloc = [&](size_t bytes) { char* r = p; p += (bytes + 255) & ~(size_t)255; return r; };
  float*    h     = (float*)   alloc(TACT * 4);  // encoder hidden (f32)
  float*    dd    = (float*)   alloc(TACT * 4);  // decoder hidden (f32)
  float*    z1f   = (float*)   alloc(TACT * 4);  // ln out f32 (encoder attn residual)
  float*    sc    = (float*)   alloc(TSC * 4);   // attention scores (f32)
  _Float16* z1h   = (_Float16*)alloc(TACT * 2);  // ln out f16
  _Float16* nkvh  = (_Float16*)alloc(TACT * 2);  // normalized encoder memory f16
  _Float16* qh    = (_Float16*)alloc(TACT * 2);
  _Float16* kh    = (_Float16*)alloc(TACT * 2);
  _Float16* vT    = (_Float16*)alloc(TACT * 2);  // V transposed (D x B*N)
  _Float16* oh    = (_Float16*)alloc(TACT * 2);  // concat heads f16
  _Float16* probs = (_Float16*)alloc(TSC * 2);   // softmax probs f16
  _Float16* midh  = (_Float16*)alloc(TMID * 2);  // FFN mid f16
  _Float16* wT    = (_Float16*)alloc(TWT * 2);   // transposed f16 weight scratch

  const long long DD = (long long)D_ * D_;
  const int R = B_ * N_;  // 2048 rows

  // ===================== encoder =====================
  embed_k<<<R, 256, 0, stream>>>(x_tokens, emb_enc, h, D_, N_);
  for (int l = 0; l < L_; l++) {
    const float* Wl = enc_W + (long long)l * 4 * DD;
    const float* lg = enc_ln_g + (long long)l * 2 * D_;
    const float* lb = enc_ln_b + (long long)l * 2 * D_;
    layernorm_k<<<R, 256, 0, stream>>>(h, lg, lb, z1h, z1f, D_);
    attention(stream, z1h, z1h, Wl, x_len, 0, /*resid=*/z1f, /*out=*/h,
              qh, kh, vT, sc, probs, oh, wT);
    layernorm_k<<<R, 256, 0, stream>>>(h, lg + D_, lb + D_, z1h, nullptr, D_);
    trcvt(stream, enc_W1 + (long long)l * D_ * 4 * D_, wT, D_, 4 * D_);
    gemmF(stream, z1h, wT, enc_b1 + (long long)l * 4 * D_, nullptr, nullptr, midh,
          R, 4 * D_, D_, D_, D_, 0, 4 * D_, 1, 1, 0, 0, 0, 0, 0, 0, 1.f, 16 | 2 | 4);
    trcvt(stream, enc_W2 + (long long)l * 4 * D_ * D_, wT, 4 * D_, D_);
    gemmF(stream, midh, wT, enc_b2 + (long long)l * D_, /*res=*/h, h, nullptr,
          R, D_, 4 * D_, 4 * D_, 4 * D_, D_, 0, 1, 1, 0, 0, 0, 0, 0, 0, 1.f, 2 | 8);
  }

  // ===================== decoder =====================
  embed_k<<<R, 256, 0, stream>>>(y_tokens, emb_dec, dd, D_, M_);
  for (int l = 0; l < L_; l++) {
    const float* mW = dec_mW + (long long)l * 4 * DD;
    const float* cW = dec_cW + (long long)l * 4 * DD;
    const float* lg = dec_ln_g + (long long)l * 3 * D_;
    const float* lb = dec_ln_b + (long long)l * 3 * D_;
    // masked self-attention (residual = d, causal mask)
    layernorm_k<<<R, 256, 0, stream>>>(dd, lg, lb, z1h, nullptr, D_);
    attention(stream, z1h, z1h, mW, nullptr, 1, /*resid=*/dd, /*out=*/dd,
              qh, kh, vT, sc, probs, oh, wT);
    // cross-attention (keys/values = ln(encoder memory), pad mask)
    layernorm_k<<<R, 256, 0, stream>>>(h,  lg + D_, lb + D_, nkvh, nullptr, D_);
    layernorm_k<<<R, 256, 0, stream>>>(dd, lg + D_, lb + D_, z1h, nullptr, D_);
    attention(stream, z1h, nkvh, cW, x_len, 0, /*resid=*/dd, /*out=*/dd,
              qh, kh, vT, sc, probs, oh, wT);
    // FFN
    layernorm_k<<<R, 256, 0, stream>>>(dd, lg + 2 * D_, lb + 2 * D_, z1h, nullptr, D_);
    trcvt(stream, dec_W1 + (long long)l * D_ * 4 * D_, wT, D_, 4 * D_);
    gemmF(stream, z1h, wT, dec_b1 + (long long)l * 4 * D_, nullptr, nullptr, midh,
          R, 4 * D_, D_, D_, D_, 0, 4 * D_, 1, 1, 0, 0, 0, 0, 0, 0, 1.f, 16 | 2 | 4);
    trcvt(stream, dec_W2 + (long long)l * 4 * D_ * D_, wT, 4 * D_, D_);
    gemmF(stream, midh, wT, dec_b2 + (long long)l * D_, /*res=*/dd, dd, nullptr,
          R, D_, 4 * D_, 4 * D_, 4 * D_, D_, 0, 1, 1, 0, 0, 0, 0, 0, 0, 1.f, 2 | 8);
  }

  // ===================== generator =====================
  layernorm_k<<<R, 256, 0, stream>>>(dd, gen_ln_g, gen_ln_b, z1h, nullptr, D_);
  trcvt(stream, gen_W, wT, D_, V_);
  gemmF(stream, z1h, wT, gen_b, nullptr, out, nullptr,
        R, V_, D_, D_, D_, V_, 0, 1, 1, 0, 0, 0, 0, 0, 0, 1.f, 2);
  log_softmax_k<<<R, 256, 0, stream>>>(out, V_);
}
